// Attention_11690900980292
// MI455X (gfx1250) — matrix operand order repaired and checked by reading, not compiled, not run on hardware
//
#include <hip/hip_runtime.h>

// ---------------------------------------------------------------------------
// Multi-head attention for MI455X (gfx1250, wave32, WMMA + async-to-LDS).
//   x[4,2048,768] @ w_qkv[768,2304] -> qkv ; flash-attn ; @ w_out[768,768]+b
// All GEMMs: v_wmma_f32_16x16x32_f16 (f16 in, f32 accumulate).
// All LDS tile staging: global_load_async_to_lds_b128 (ASYNCcnt), double-
// buffered. Weights and V are pre-transposed once so every staging copy is a
// straight 16B move and every operand fragment is two contiguous
// ds_load_b128 (no repack VALU, no strided LDS gathers).
// ---------------------------------------------------------------------------

typedef _Float16 half_t;
typedef __attribute__((ext_vector_type(16))) _Float16 v16h;
typedef __attribute__((ext_vector_type(8)))  float    v8f;

union AFrag { v16h v; half_t h[16]; uint4 q[2]; };
union BFrag { v16h v; half_t h[16]; uint4 q[2]; };
union CFrag { v8f  v; float  f[8];  };

// ISA 7.12.2: 16-bit A-matrix 16x32 K map (unused; the Q load uses contiguous k = 16*hi + h)
__device__ __forceinline__ int kmapA(int h, int hi) {
    return (h & 7) + ((h & 8) << 1) + (hi << 3);
}
// destination position of source K-group g (0..3) in a permuted A-layout row:
// g0->0, g1->2, g2->1, g3->3  (2-bit bit-swap)
__device__ __forceinline__ int dstgrp(int g) {
    return ((g & 1) << 1) | (g >> 1);
}

// Async 16B global->LDS copy (GVS mode: SGPR64 base + per-lane i32 offset).
// vdst VGPR carries the LDS byte address (= low 32 bits of generic pointer).
__device__ __forceinline__ void async_b128(half_t* lds_dst,
                                           const half_t* gbase,
                                           uint32_t byte_off) {
    uint32_t ldsa = (uint32_t)(uintptr_t)lds_dst;
    asm volatile("global_load_async_to_lds_b128 %0, %1, %2"
                 :
                 : "v"(ldsa), "v"(byte_off), "s"(gbase)
                 : "memory");
}
__device__ __forceinline__ void wait_async0() {
    asm volatile("s_wait_asynccnt 0x0" ::: "memory");
}

#define WMMA(a, b, c) \
    __builtin_amdgcn_wmma_f32_16x16x32_f16(false, (a), false, (b), (short)0, (c), false, false)

// Problem constants
#define SEQ   2048
#define EMBD  768
#define HEADS 12
#define HDIM  64
#define C3    2304          // 3*EMBD
#define MROWS 8192          // B*S

// ---------------------------------------------------------------------------
__global__ void f32_to_f16_kernel(const float* __restrict__ src,
                                  half_t* __restrict__ dst, int n) {
    int i = blockIdx.x * blockDim.x + threadIdx.x;
    if (i < n) dst[i] = (half_t)src[i];
}

// src [R][C] f32  ->  dst [C][R] f16   (one-time weight transpose)
__global__ void transpose_f32_to_f16_kernel(const float* __restrict__ src,
                                            half_t* __restrict__ dst,
                                            int R, int C) {
    int idx = blockIdx.x * blockDim.x + threadIdx.x;
    if (idx < R * C) {
        int c = idx / R, r = idx % R;          // writes coalesced over dst
        dst[idx] = (half_t)src[(size_t)r * C + c];
    }
}

// vt[b][h][d][s] = qkv[b][s][2*EMBD + h*64 + d]   (one-time V transpose)
__global__ void vtranspose_kernel(const half_t* __restrict__ qkv,
                                  half_t* __restrict__ vt) {
    int idx = blockIdx.x * blockDim.x + threadIdx.x;   // over B*H*D*S
    int s  = idx & (SEQ - 1);
    int t  = idx >> 11;
    int d  = t & (HDIM - 1);
    int hb = t >> 6;                   // b*HEADS + h
    int h  = hb % HEADS, b = hb / HEADS;
    vt[idx] = qkv[((size_t)b * SEQ + s) * C3 + 2 * EMBD + h * HDIM + d];
}

// ---------------------------------------------------------------------------
// Tiled GEMM: C[M,N] = A[M,K] * B[K,N] with B supplied TRANSPOSED (BT [N][K]).
// Block = 256 threads (8 waves), tile 128x128, k-step 32, double-buffered
// async-to-LDS staging. Waves 4x2; each wave owns 32x64 = 2x4 WMMA C tiles.
// OUT_F16==1: store f16 (qkv). OUT_F16==0: store f32 + bias (final proj).
// ---------------------------------------------------------------------------
template <int OUT_F16>
__global__ __launch_bounds__(256)
void gemm_wmma_kernel(const half_t* __restrict__ A, const half_t* __restrict__ BT,
                      half_t* __restrict__ Ch, float* __restrict__ Cf,
                      const float* __restrict__ bias, int M, int N, int K) {
    __shared__ half_t Atile[2][128][48];   // K-group-permuted rows; 96B stride
    __shared__ half_t Btile[2][128][48];   // [n][k] rows;           96B stride

    const int tid  = threadIdx.x;
    const int wave = tid >> 5, lane = tid & 31;
    const int lr   = lane & 15, hi = lane >> 4;
    const int waveM = wave >> 1;        // 0..3  (32 rows each)
    const int waveN = wave & 1;         // 0..1  (64 cols each)
    const int rowBase = blockIdx.y * 128;
    const int colBase = blockIdx.x * 128;

    CFrag acc[2][4] = {};

    auto stage = [&](int buf, int k0) {
#pragma unroll
        for (int c = tid; c < 512; c += 256) {      // A: 128 rows x 4 groups
            int r = c >> 2, g = c & 3;
            async_b128(&Atile[buf][r][dstgrp(g) * 8], A,
                       (uint32_t)((((size_t)(rowBase + r)) * K + k0 + g * 8) * 2));
        }
#pragma unroll
        for (int c = tid; c < 512; c += 256) {      // BT: 128 cols x 4 groups
            int n = c >> 2, g = c & 3;
            async_b128(&Btile[buf][n][dstgrp(g) * 8], BT,
                       (uint32_t)((((size_t)(colBase + n)) * K + k0 + g * 8) * 2));
        }
    };

    const int nk = K / 32;
    stage(0, 0);
    for (int it = 0; it < nk; ++it) {
        int cur = it & 1;
        wait_async0();                 // tile `it` fully in LDS (this wave)
        __syncthreads();               // visible to all; tile it-1 compute done
        if (it + 1 < nk) stage(cur ^ 1, (it + 1) * 32);

        AFrag af[2];
        BFrag bf[4];
#pragma unroll
        for (int i = 0; i < 2; ++i) {
            const uint4* p = (const uint4*)&Atile[cur][waveM * 32 + i * 16 + lr][hi * 16];
            af[i].q[0] = p[0];
            af[i].q[1] = p[1];
        }
#pragma unroll
        for (int j = 0; j < 4; ++j) {
            const uint4* p = (const uint4*)&Btile[cur][waveN * 64 + j * 16 + lr][hi * 16];
            bf[j].q[0] = p[0];
            bf[j].q[1] = p[1];
        }
#pragma unroll
        for (int i = 0; i < 2; ++i)
#pragma unroll
            for (int j = 0; j < 4; ++j)
                acc[i][j].v = WMMA(af[i].v, bf[j].v, acc[i][j].v);
    }

    // Epilogue. C layout: row = e + 8*hi within tile, col = lr within tile.
#pragma unroll
    for (int i = 0; i < 2; ++i)
#pragma unroll
        for (int j = 0; j < 4; ++j)
#pragma unroll
            for (int e = 0; e < 8; ++e) {
                int m = rowBase + waveM * 32 + i * 16 + e + 8 * hi;
                int n = colBase + waveN * 64 + j * 16 + lr;
                float v = acc[i][j].f[e];
                if (OUT_F16)
                    Ch[(size_t)m * N + n] = (half_t)v;
                else
                    Cf[(size_t)m * N + n] = v + bias[n];
            }
}

// ---------------------------------------------------------------------------
// Fused flash attention. Q,K from qkv f16 [B,S,2304]; V from pre-transposed
// vt f16 [B,H,64,S]. Block = 8 waves; each wave owns 16 query rows (block:
// 128 rows of one (b,h)). Keys in chunks of 32, K/V double-buffered via
// async-to-LDS:
//  - Kt[key][d]  : QK^T B-fragment reads contiguous in d.
//  - Vtt[d][key] : PV B-fragment reads contiguous in key (straight copy now).
//  - Pt          : per-wave P staging as plain [row][key] (contiguous reads).
// Online softmax in registers; row stats via 16-lane shfl_xor butterflies
// (C-fragment rows map to vgpr index + 8*(lane>=16)).
// ---------------------------------------------------------------------------
__global__ __launch_bounds__(256)
void flash_attn_kernel(const half_t* __restrict__ qkv,
                       const half_t* __restrict__ vt,
                       half_t* __restrict__ attn) {
    const int bh = blockIdx.y;
    const int b  = bh / HEADS, head = bh % HEADS;
    const int tid  = threadIdx.x;
    const int wave = tid >> 5, lane = tid & 31;
    const int lr   = lane & 15, hi = lane >> 4;
    const int qbase = blockIdx.x * 128 + wave * 16;
    const half_t* base = qkv + (size_t)b * SEQ * C3;

    __shared__ half_t Kt[2][32][72];     // [key][d] (+8 pad), 144B rows
    __shared__ half_t Vtt[2][64][48];    // [d][key], 96B rows
    __shared__ half_t Pt[8][16][48];     // per-wave P, [row][key]

    // Q fragments (16 rows x 64 dims = two 16x32 A fragments), loaded once.
    AFrag qf[2];
#pragma unroll
    for (int f = 0; f < 2; ++f)
#pragma unroll
        for (int h = 0; h < 16; ++h) {
            int k = f * 32 + hi * 16 + h;
            qf[f].h[h] = base[(size_t)(qbase + lr) * C3 + head * HDIM + k];
        }

    auto stage_kv = [&](int buf, int kb) {
        // K chunk: 32 keys x 64 d = 256 16B chunks, 1 per thread
        int r = tid >> 3, w8 = (tid & 7) * 8;
        async_b128(&Kt[buf][r][w8], qkv,
                   (uint32_t)(((((size_t)b * SEQ + kb + r) * C3) +
                               EMBD + head * HDIM + w8) * 2));
        // V chunk from vt[b][h][d][s]: 64 d-rows x 32 keys, 1 chunk per thread
        int d = tid >> 2, g = tid & 3;
        async_b128(&Vtt[buf][d][g * 8], vt,
                   (uint32_t)(((((size_t)b * HEADS + head) * HDIM + d) * SEQ +
                               kb + g * 8) * 2));
    };

    CFrag o[4] = {};                      // 16 rows x 64 dims accumulator
    float mst[8], lst[8];
#pragma unroll
    for (int e = 0; e < 8; ++e) { mst[e] = -1e30f; lst[e] = 0.0f; }

    const int nchunks = SEQ / 32;
    stage_kv(0, 0);
    for (int it = 0; it < nchunks; ++it) {
        int cur = it & 1;
        wait_async0();
        __syncthreads();
        if (it + 1 < nchunks) stage_kv(cur ^ 1, (it + 1) * 32);

        // S = (Q K^T) / sqrt(64): two 16x16 tiles (keys 0-15, 16-31)
        CFrag s[2];
#pragma unroll
        for (int t = 0; t < 2; ++t) {
            BFrag bk0, bk1;               // B = K^T: col = key, K-dim = d
            const uint4* p = (const uint4*)&Kt[cur][t * 16 + lr][hi * 16];
            bk0.q[0] = p[0];  bk0.q[1] = p[1];   // d 0..31 slice
            bk1.q[0] = p[4];  bk1.q[1] = p[5];   // d 32..63 slice
            v8f z = {};
            s[t].v = WMMA(qf[0].v, bk0.v, z);
            s[t].v = WMMA(qf[1].v, bk1.v, s[t].v);
#pragma unroll
            for (int e = 0; e < 8; ++e) s[t].f[e] *= 0.125f;
        }

        // row max across 32 keys: elementwise over tiles, then 16-lane butterfly
        float mx[8];
#pragma unroll
        for (int e = 0; e < 8; ++e) mx[e] = fmaxf(s[0].f[e], s[1].f[e]);
#pragma unroll
        for (int msk = 1; msk < 16; msk <<= 1)
#pragma unroll
            for (int e = 0; e < 8; ++e)
                mx[e] = fmaxf(mx[e], __shfl_xor(mx[e], msk, 16));

        float alpha[8];
#pragma unroll
        for (int e = 0; e < 8; ++e) {
            float mn = fmaxf(mst[e], mx[e]);
            alpha[e] = __expf(mst[e] - mn);
            mst[e]   = mn;
        }
        // P = exp(S - m_new); row sums
#pragma unroll
        for (int t = 0; t < 2; ++t)
#pragma unroll
            for (int e = 0; e < 8; ++e) s[t].f[e] = __expf(s[t].f[e] - mst[e]);
        float rs[8];
#pragma unroll
        for (int e = 0; e < 8; ++e) rs[e] = s[0].f[e] + s[1].f[e];
#pragma unroll
        for (int msk = 1; msk < 16; msk <<= 1)
#pragma unroll
            for (int e = 0; e < 8; ++e) rs[e] += __shfl_xor(rs[e], msk, 16);
#pragma unroll
        for (int e = 0; e < 8; ++e) lst[e] = lst[e] * alpha[e] + rs[e];
        // rescale running output
#pragma unroll
        for (int tj = 0; tj < 4; ++tj)
#pragma unroll
            for (int e = 0; e < 8; ++e) o[tj].f[e] *= alpha[e];

        // P: C layout -> A layout via per-wave LDS, stored [row][key]
        // (intra-wave DS ops are in-order, no barrier needed)
#pragma unroll
        for (int t = 0; t < 2; ++t) {
            int key = t * 16 + lr;
            int pos = key;
#pragma unroll
            for (int e = 0; e < 8; ++e)
                Pt[wave][e + 8 * hi][pos] = (half_t)s[t].f[e];
        }
        AFrag pf;
        {
            const uint4* p = (const uint4*)&Pt[wave][lr][hi * 16];
            pf.q[0] = p[0];
            pf.q[1] = p[1];
        }

        // O += P(16x32) * V(32x64): 4 WMMAs, V fragments contiguous from Vtt
#pragma unroll
        for (int tj = 0; tj < 4; ++tj) {
            BFrag bv;
            const uint4* p = (const uint4*)&Vtt[cur][tj * 16 + lr][hi * 16];
            bv.q[0] = p[0];
            bv.q[1] = p[1];
            o[tj].v = WMMA(pf.v, bv.v, o[tj].v);
        }
    }

    // normalize and write attn[b, s, head*64 + d] as f16
    float inv[8];
#pragma unroll
    for (int e = 0; e < 8; ++e) inv[e] = 1.0f / lst[e];
#pragma unroll
    for (int tj = 0; tj < 4; ++tj)
#pragma unroll
        for (int e = 0; e < 8; ++e) {
            int row = qbase + e + 8 * hi;
            int col = head * HDIM + tj * 16 + lr;
            attn[((size_t)b * SEQ + row) * EMBD + col] = (half_t)(o[tj].f[e] * inv[e]);
        }
}

// ---------------------------------------------------------------------------
extern "C" void kernel_launch(void* const* d_in, const int* in_sizes, int n_in,
                              void* d_out, int out_size, void* d_ws, size_t ws_size,
                              hipStream_t stream) {
    const float* x     = (const float*)d_in[0];   // [4,2048,768]
    const float* w_qkv = (const float*)d_in[1];   // [768,2304]
    const float* w_out = (const float*)d_in[2];   // [768,768]
    const float* b_out = (const float*)d_in[3];   // [768]
    float* out = (float*)d_out;                   // [4,2048,768] f32

    // workspace layout (all segments naturally 256B-aligned)
    char* ws = (char*)d_ws;
    size_t ofs = 0;
    half_t* x_h     = (half_t*)(ws + ofs); ofs += (size_t)MROWS * EMBD * 2;  // 12.6 MB
    half_t* wqkvT_h = (half_t*)(ws + ofs); ofs += (size_t)EMBD * C3 * 2;     //  3.5 MB
    half_t* woutT_h = (half_t*)(ws + ofs); ofs += (size_t)EMBD * EMBD * 2;   //  1.2 MB
    half_t* qkv_h   = (half_t*)(ws + ofs); ofs += (size_t)MROWS * C3 * 2;    // 37.7 MB
    half_t* vt_h    = (half_t*)(ws + ofs); ofs += (size_t)MROWS * EMBD * 2;  // 12.6 MB
    half_t* attn_h  = (half_t*)(ws + ofs); ofs += (size_t)MROWS * EMBD * 2;  // 12.6 MB
    (void)ws_size;

    // conversions: x straight; weights transposed to [N][K]
    {
        int n = MROWS * EMBD;
        f32_to_f16_kernel<<<(n + 255) / 256, 256, 0, stream>>>(x, x_h, n);
        n = EMBD * C3;
        transpose_f32_to_f16_kernel<<<(n + 255) / 256, 256, 0, stream>>>(
            w_qkv, wqkvT_h, EMBD, C3);
        n = EMBD * EMBD;
        transpose_f32_to_f16_kernel<<<(n + 255) / 256, 256, 0, stream>>>(
            w_out, woutT_h, EMBD, EMBD);
    }

    // QKV projection: [8192,768] x [768,2304] -> qkv f16
    gemm_wmma_kernel<1><<<dim3(C3 / 128, MROWS / 128), 256, 0, stream>>>(
        x_h, wqkvT_h, qkv_h, nullptr, nullptr, MROWS, C3, EMBD);

    // one-time V transpose: vt[b][h][d][s]
    {
        int n = MROWS * EMBD;
        vtranspose_kernel<<<(n + 255) / 256, 256, 0, stream>>>(qkv_h, vt_h);
    }

    // fused flash attention -> attn f16 [8192,768]
    flash_attn_kernel<<<dim3(SEQ / 128, 4 * HEADS), 256, 0, stream>>>(
        qkv_h, vt_h, attn_h);

    // output projection + bias: [8192,768] x [768,768] -> f32 out
    gemm_wmma_kernel<0><<<dim3(EMBD / 128, MROWS / 128), 256, 0, stream>>>(
        attn_h, woutT_h, nullptr, out, b_out, MROWS, EMBD, EMBD);
}
